// DifferentiableRAM_5248450036349
// MI455X (gfx1250) — matrix operand order repaired and checked by reading, not compiled, not run on hardware
//
#include <hip/hip_runtime.h>
#include <hip/hip_bf16.h>

typedef __attribute__((ext_vector_type(16))) __bf16 v16bf;
typedef __attribute__((ext_vector_type(8)))  float  v8f;

#define B_SZ   32
#define C_SZ   3
#define HW_SZ  512   // H == W == 512
#define N_G    256   // glimpse points
#define K_D    512
#define SMALLF 1e-4f

// Workspace layout (bytes):
//   Fy  bf16 [32][256][512] : 8,388,608   @ 0
//   Fx  bf16 [32][256][512] : 8,388,608   @ 8 MiB
//   Fyx bf16 [96][256][512] : 25,165,824  @ 16 MiB
#define WS_FY_OFF   0
#define WS_FX_OFF   (8u  * 1024u * 1024u)
#define WS_FYX_OFF  (16u * 1024u * 1024u)

// ---------------------------------------------------------------------------
// Kernel 1: Gaussian filterbanks, one wave32 per (matrix, b, n) row.
// ---------------------------------------------------------------------------
__global__ void __launch_bounds__(256)
draw_fb_kernel(const float* __restrict__ p,
               __bf16* __restrict__ Fy,
               __bf16* __restrict__ Fx) {
    const int wave = threadIdx.x >> 5;
    const int lane = threadIdx.x & 31;
    int rid = blockIdx.x * 8 + wave;           // 0..16383
    const int isFx = rid >> 13;                // 0 = Fy, 1 = Fx
    rid &= 8191;
    const int b = rid >> 8;
    const int n = rid & 255;

    const float* pb = p + b * 5;
    const float sigma2 = __expf(pb[2]);
    const float delta  = __expf(pb[3]) * (511.0f / 255.0f);
    const float g = isFx ? ((float)HW_SZ * (pb[0] + 1.0f) * 0.5f)
                         : ((float)HW_SZ * (pb[1] + 1.0f) * 0.5f);
    const float mu     = g + delta * ((float)n - 128.5f);
    const float nhalf  = -0.5f / sigma2;

    const int abase = lane * 16;               // 16 consecutive 'a' per lane
    float vals[16];
    float sum = 0.0f;
#pragma unroll
    for (int j = 0; j < 16; ++j) {
        float d = (float)(abase + j) - mu;
        float e = __expf(d * d * nhalf);
        vals[j] = e;
        sum += e;
    }
#pragma unroll
    for (int off = 16; off > 0; off >>= 1)
        sum += __shfl_xor(sum, off, 32);
    const float sc = 1.0f / (sum + SMALLF);

    union { __bf16 h[16]; uint4 q[2]; } ob;
#pragma unroll
    for (int j = 0; j < 16; ++j)
        ob.h[j] = (__bf16)(vals[j] * sc);

    __bf16* dst = (isFx ? Fx : Fy) + (size_t)rid * HW_SZ + abase;
    uint4* dq = (uint4*)dst;
    dq[0] = ob.q[0];
    dq[1] = ob.q[1];
}

// ---------------------------------------------------------------------------
// WMMA fragment loaders (bf16, 16x16x32) from LDS tiles stored [row][64 k]
// bf16, viewed as 32-bit words (two consecutive k per word; row = 32 words).
// kw = 32-bit-word offset of the 32-k slice within the row (0 or 16).
// A layout (ISA 7.12.2): lane -> M, VGPR v -> K = (v>=4?16:0)+half*8+(v&3)*2
// B layout:              lane -> N, VGPR v -> K = (v>=4?16:0)+half*8+(v&3)*2
// ---------------------------------------------------------------------------
union Frag { unsigned int u[8]; v16bf v; };

__device__ __forceinline__ void load_a_frag(const unsigned int* As32, int mbase,
                                            int kw, int lane, Frag& f) {
    const int h = lane >> 4;
    const unsigned int* base = As32 + (mbase + (lane & 15)) * 32 + kw + h * 4;
#pragma unroll
    for (int v = 0; v < 8; ++v)
        f.u[v] = base[((v & 4) ? 8 : 0) + (v & 3)];
}

__device__ __forceinline__ void load_b_frag(const unsigned int* Bs32, int nbase,
                                            int kw, int lane, Frag& f) {
    const int h = lane >> 4;
    const unsigned int* base = Bs32 + (nbase + (lane & 15)) * 32 + kw + h * 4;
#pragma unroll
    for (int v = 0; v < 8; ++v)
        f.u[v] = base[((v & 4) ? 8 : 0) + (v & 3)];
}

#define WMMA_BF16(A, B, C) \
    __builtin_amdgcn_wmma_f32_16x16x32_bf16(false, (A).v, false, (B).v, (short)0, (C), false, false)

// ---------------------------------------------------------------------------
// Kernel 2: Fyx[bc] = Fy[b] (256x512 bf16) @ x[bc] (512x512 f32->bf16)
// WG tile 128(M) x 64(W), BK=64; 8 waves as 4(M) x 2(N), 32x32 per wave.
// ---------------------------------------------------------------------------
__global__ void __launch_bounds__(256)
draw_gemm1_kernel(const float* __restrict__ x,
                  const __bf16* __restrict__ Fy,
                  __bf16* __restrict__ Fyx) {
    __shared__ __align__(16) __bf16 As[128 * 64];  // [m][k]   16 KB
    __shared__ __align__(16) __bf16 Bs[64 * 64];   // [w][k]    8 KB (x transposed)

    const int tid  = threadIdx.x;
    const int lane = tid & 31;
    const int wave = tid >> 5;
    const int wm   = wave >> 1;      // 0..3 -> m offset wm*32
    const int wn   = wave & 1;       // 0..1 -> n offset wn*32

    const int bc = blockIdx.z;       // b*3 + c
    const int b  = bc / 3;
    const int m0 = blockIdx.y * 128;
    const int w0 = blockIdx.x * 64;

    const __bf16* Arow = Fy + (size_t)(b * N_G + m0) * K_D;
    const float*  Xb   = x  + (size_t)bc * HW_SZ * HW_SZ;

    v8f acc[2][2];
    v8f zero = {};
    acc[0][0] = zero; acc[0][1] = zero; acc[1][0] = zero; acc[1][1] = zero;

    const unsigned int* As32 = (const unsigned int*)As;
    unsigned int*       Bs32 = (unsigned int*)Bs;

    // staging index precompute
    const int a_row = tid >> 1;               // 0..127
    const int a_kh  = (tid & 1) * 32;         // 0 / 32
    const int b_wl  = tid & 63;               // 0..63
    const int b_kb  = (tid >> 6) * 16;        // 0/16/32/48

    for (int k0 = 0; k0 < K_D; k0 += 64) {
        // prefetch next k-step sources (global_prefetch_b8)
        if (k0 + 64 < K_D) {
            __builtin_prefetch(Arow + (size_t)a_row * K_D + k0 + 64 + a_kh, 0, 0);
            __builtin_prefetch(&Xb[(size_t)(k0 + 64 + b_kb) * HW_SZ + w0 + b_wl], 0, 0);
        }
        // --- stage A tile: 128 rows x 64 k (bf16 row-major copy, 4x uint4) ---
        {
            const uint4* src = (const uint4*)(Arow + (size_t)a_row * K_D + k0 + a_kh);
            uint4* dst = (uint4*)(As + a_row * 64 + a_kh);
            dst[0] = src[0]; dst[1] = src[1]; dst[2] = src[2]; dst[3] = src[3];
        }
        // --- stage B tile: x[k0..k0+63][w0..w0+63], f32 -> packed bf16x2,
        //     transposed to Bs[w][k]; one 32-bit LDS store per k-pair ---
#pragma unroll
        for (int j = 0; j < 8; ++j) {
            const int k = b_kb + 2 * j;
            const float f0 = Xb[(size_t)(k0 + k)     * HW_SZ + w0 + b_wl];
            const float f1 = Xb[(size_t)(k0 + k + 1) * HW_SZ + w0 + b_wl];
            union { __bf16 h[2]; unsigned int u; } pk;
            pk.h[0] = (__bf16)f0;
            pk.h[1] = (__bf16)f1;
            Bs32[b_wl * 32 + (b_kb >> 1) + j] = pk.u;
        }
        __syncthreads();

#pragma unroll
        for (int ks = 0; ks < 2; ++ks) {       // two 32-k slices
            const int kw = ks * 16;
            Frag a0, a1, bb0, bb1;
            load_a_frag(As32, wm * 32 + 0,  kw, lane, a0);
            load_a_frag(As32, wm * 32 + 16, kw, lane, a1);
            load_b_frag(Bs32, wn * 32 + 0,  kw, lane, bb0);
            load_b_frag(Bs32, wn * 32 + 16, kw, lane, bb1);
            acc[0][0] = WMMA_BF16(a0, bb0, acc[0][0]);
            acc[0][1] = WMMA_BF16(a0, bb1, acc[0][1]);
            acc[1][0] = WMMA_BF16(a1, bb0, acc[1][0]);
            acc[1][1] = WMMA_BF16(a1, bb1, acc[1][1]);
        }
        __syncthreads();
    }

    // store bf16 Fyx; C/D layout: VGPR r -> M = half*8 + r, lane -> N
    const int h  = lane >> 4;
    const int cl = lane & 15;
    __bf16* Obase = Fyx + (size_t)bc * N_G * HW_SZ;
#pragma unroll
    for (int mi = 0; mi < 2; ++mi)
#pragma unroll
        for (int ni = 0; ni < 2; ++ni) {
            const int mbase = m0 + wm * 32 + mi * 16 + h * 8;
            const int nbase = w0 + wn * 32 + ni * 16 + cl;
#pragma unroll
            for (int r = 0; r < 8; ++r)
                Obase[(size_t)(mbase + r) * HW_SZ + nbase] = (__bf16)acc[mi][ni][r];
        }
}

// ---------------------------------------------------------------------------
// Kernel 3: out[bc] = gamma[b] * Fyx[bc] (256x512) @ Fx[b]^T (512x256)
// ---------------------------------------------------------------------------
__global__ void __launch_bounds__(256)
draw_gemm2_kernel(const float* __restrict__ p,
                  const __bf16* __restrict__ Fyx,
                  const __bf16* __restrict__ Fx,
                  float* __restrict__ out) {
    __shared__ __align__(16) __bf16 As[128 * 64];  // [n_row][k] 16 KB
    __shared__ __align__(16) __bf16 Bs[64 * 64];   // [m_col][k]  8 KB

    const int tid  = threadIdx.x;
    const int lane = tid & 31;
    const int wave = tid >> 5;
    const int wm   = wave >> 1;
    const int wn   = wave & 1;

    const int bc = blockIdx.z;
    const int b  = bc / 3;
    const int m0 = blockIdx.y * 128;   // output rows  (n in reference)
    const int n0 = blockIdx.x * 64;    // output cols  (m in reference)

    const __bf16* Arow = Fyx + (size_t)(bc * N_G + m0) * K_D;
    const __bf16* Brow = Fx  + (size_t)(b  * N_G + n0) * K_D;
    const float gamma = __expf(p[b * 5 + 4]);

    v8f acc[2][2];
    v8f zero = {};
    acc[0][0] = zero; acc[0][1] = zero; acc[1][0] = zero; acc[1][1] = zero;

    const unsigned int* As32 = (const unsigned int*)As;
    const unsigned int* Bs32 = (const unsigned int*)Bs;

    const int a_row = tid >> 1;
    const int a_kh  = (tid & 1) * 32;
    const int b_row = tid >> 2;
    const int b_seg = (tid & 3) * 16;

    for (int k0 = 0; k0 < K_D; k0 += 64) {
        if (k0 + 64 < K_D) {
            __builtin_prefetch(Arow + (size_t)a_row * K_D + k0 + 64 + a_kh, 0, 0);
            __builtin_prefetch(Brow + (size_t)b_row * K_D + k0 + 64 + b_seg, 0, 0);
        }
        {   // A tile: 128 x 64 bf16
            const uint4* src = (const uint4*)(Arow + (size_t)a_row * K_D + k0 + a_kh);
            uint4* dst = (uint4*)(As + a_row * 64 + a_kh);
            dst[0] = src[0]; dst[1] = src[1]; dst[2] = src[2]; dst[3] = src[3];
        }
        {   // B tile: 64 Fx rows x 64 k, straight coalesced copy
            const uint4* src = (const uint4*)(Brow + (size_t)b_row * K_D + k0 + b_seg);
            uint4* dst = (uint4*)(Bs + b_row * 64 + b_seg);
            dst[0] = src[0]; dst[1] = src[1];
        }
        __syncthreads();

#pragma unroll
        for (int ks = 0; ks < 2; ++ks) {
            const int kw = ks * 16;
            Frag a0, a1, bb0, bb1;
            load_a_frag(As32, wm * 32 + 0,  kw, lane, a0);
            load_a_frag(As32, wm * 32 + 16, kw, lane, a1);
            load_b_frag(Bs32, wn * 32 + 0,  kw, lane, bb0);
            load_b_frag(Bs32, wn * 32 + 16, kw, lane, bb1);
            acc[0][0] = WMMA_BF16(a0, bb0, acc[0][0]);
            acc[0][1] = WMMA_BF16(a0, bb1, acc[0][1]);
            acc[1][0] = WMMA_BF16(a1, bb0, acc[1][0]);
            acc[1][1] = WMMA_BF16(a1, bb1, acc[1][1]);
        }
        __syncthreads();
    }

    const int h  = lane >> 4;
    const int cl = lane & 15;
    float* Obase = out + (size_t)bc * N_G * N_G;
#pragma unroll
    for (int mi = 0; mi < 2; ++mi)
#pragma unroll
        for (int ni = 0; ni < 2; ++ni) {
            const int mbase = m0 + wm * 32 + mi * 16 + h * 8;
            const int nbase = n0 + wn * 32 + ni * 16 + cl;
#pragma unroll
            for (int r = 0; r < 8; ++r)
                Obase[(size_t)(mbase + r) * N_G + nbase] = gamma * acc[mi][ni][r];
        }
}

// ---------------------------------------------------------------------------
extern "C" void kernel_launch(void* const* d_in, const int* in_sizes, int n_in,
                              void* d_out, int out_size, void* d_ws, size_t ws_size,
                              hipStream_t stream) {
    const float* x = (const float*)d_in[0];   // [32,3,512,512] f32
    const float* p = (const float*)d_in[1];   // [32,5] f32

    char* ws = (char*)d_ws;
    __bf16* Fy  = (__bf16*)(ws + WS_FY_OFF);
    __bf16* Fx  = (__bf16*)(ws + WS_FX_OFF);
    __bf16* Fyx = (__bf16*)(ws + WS_FYX_OFF);

    // 16384 filterbank rows (Fy + Fx), 8 waves per block
    draw_fb_kernel<<<2048, 256, 0, stream>>>(p, Fy, Fx);

    // GEMM1: grid = (W/64, M/128, B*C)
    draw_gemm1_kernel<<<dim3(8, 2, 96), 256, 0, stream>>>(x, Fy, Fyx);

    // GEMM2: grid = (Ncols/64, M/128, B*C)
    draw_gemm2_kernel<<<dim3(4, 2, 96), 256, 0, stream>>>(p, Fyx, Fx, (float*)d_out);
}
